// Attention_17635135717786
// MI455X (gfx1250) — matrix operand order repaired and checked by reading, not compiled, not run on hardware
//
#include <hip/hip_runtime.h>
#include <hip/hip_bf16.h>

// ---------------------------------------------------------------------------
// Attention block for MI455X (gfx1250, wave32, WMMA).
// All GEMMs use v_wmma_f32_16x16x32_f16 (f16 in, f32 accumulate).
// GEMMs are register-blocked 32x64 per wave (8 WMMAs per 12 b128 loads).
// Attention is fused flash-style: S tiles in registers, P restaged via LDS
// (C-layout -> A-layout), online softmax, never spills N x N to HBM.
// ---------------------------------------------------------------------------

typedef __attribute__((ext_vector_type(16))) _Float16 v16h;
typedef __attribute__((ext_vector_type(8)))  _Float16 v8h;
typedef __attribute__((ext_vector_type(8)))  float    v8f;

#define BN_   32
#define DIM_  256
#define FM_   32
#define NP_   1024           // tokens per image
#define H_    8
#define DK_   32
#define DV_   64
#define OQKV_ 1024           // 256 q + 256 k + 512 v channels
#define SCALE_     0.17677669529663687f   // 1/sqrt(32)
#define INV_SCALE_ 5.656854249492381f     // sqrt(32)

union FragU { v16h v; v8h h[2]; };

__device__ __forceinline__ float gelu_exact(float x) {
    return 0.5f * x * (1.0f + erff(x * 0.70710678118654752f));
}

// --------------------------- prep kernels ----------------------------------

// Pack Wq|Wk|Wv -> f16 [1024,256] row-major, Wo -> f16 [256,512] row-major.
__global__ __launch_bounds__(256) void convert_weights_kernel(
    const float* __restrict__ Wq, const float* __restrict__ Wk,
    const float* __restrict__ Wv, const float* __restrict__ Wo,
    _Float16* __restrict__ wqkv, _Float16* __restrict__ wo) {
    int id = blockIdx.x * blockDim.x + threadIdx.x;
    const int n_qkv = OQKV_ * DIM_;          // 262144
    const int n_o   = DIM_ * (H_ * DV_);     // 131072
    if (id < n_qkv) {
        int o = id / DIM_, c = id % DIM_;
        float v;
        if (o < 256)      v = Wq[o * DIM_ + c];
        else if (o < 512) v = Wk[(o - 256) * DIM_ + c];
        else              v = Wv[(o - 512) * DIM_ + c];
        wqkv[id] = (_Float16)v;
    } else if (id < n_qkv + n_o) {
        int i = id - n_qkv;
        wo[i] = (_Float16)Wo[i];
    }
}

// x [b][c][n] f32 -> xT [b][n][c] f16  (B-operand layout for QKV GEMM)
__global__ __launch_bounds__(256) void transpose_x_kernel(
    const float* __restrict__ x, _Float16* __restrict__ xt) {
    int id = blockIdx.x * blockDim.x + threadIdx.x;   // B*DIM*N
    int n = id % NP_;
    int c = (id / NP_) % DIM_;
    int b = id / (NP_ * DIM_);
    xt[((size_t)b * NP_ + n) * DIM_ + c] = (_Float16)x[id];
}

// bias[h][i][j] = pos_table[pos_indices[i*N+j]*H + h] * sqrt(dk)
__global__ __launch_bounds__(256) void build_bias_kernel(
    const float* __restrict__ pos_table, const int* __restrict__ pos_idx,
    float* __restrict__ biasb) {
    int id = blockIdx.x * blockDim.x + threadIdx.x;   // H * N * N
    int r = id & (NP_ * NP_ - 1);
    int h = id >> 20;                                  // N*N = 2^20
    biasb[id] = pos_table[pos_idx[r] * H_ + h] * INV_SCALE_;
}

// --------------------------- generic WMMA GEMM -----------------------------
// C[b][m][n] = sum_k A[m][k] * BT[b][n][k]  (+ bias[m]).
// Register-blocked: one wave owns a 32 (M) x 64 (N) macro-tile:
// per K-step (32): 2 A-frags + 4 B-frags (12 x b128) -> 8 WMMAs.
__global__ __launch_bounds__(256) void wmma_gemm_kernel(
    const _Float16* __restrict__ A, const _Float16* __restrict__ BT,
    float* __restrict__ C, const float* __restrict__ bias,
    int M, int Nn, int K, long bStride, long cStride, int tilesPerBatch) {
    const int lane = threadIdx.x & 31;
    const int half = lane >> 4, col = lane & 15;
    int t = blockIdx.x * (blockDim.x >> 5) + (threadIdx.x >> 5);
    int b  = t / tilesPerBatch;
    int rr = t % tilesPerBatch;
    int ntiles = Nn >> 6;                 // 64-wide macro tiles along N
    int tm = rr / ntiles, tn = rr % ntiles;

    const _Float16* Bb = BT + (long)b * bStride;
    float* Cb = C + (long)b * cStride;

    const int am0 = tm * 32 + col;        // A rows owned by this lane
    const int bn0 = tn * 64 + col;        // B cols owned by this lane

    v8f acc[2][4] = {};
    for (int k0 = 0; k0 < K; k0 += 32) {
        FragU a[2];
        #pragma unroll
        for (int i = 0; i < 2; ++i) {
            const _Float16* ar = A + (long)(am0 + i * 16) * K + k0;
            a[i].h[0] = *(const v8h*)(ar + half * 16);
            a[i].h[1] = *(const v8h*)(ar + half * 16 + 8);
        }
        FragU bf[4];
        #pragma unroll
        for (int j = 0; j < 4; ++j) {
            const _Float16* br = Bb + (long)(bn0 + j * 16) * K + k0 + half * 16;
            bf[j].h[0] = *(const v8h*)(br);
            bf[j].h[1] = *(const v8h*)(br + 8);
        }
        #pragma unroll
        for (int i = 0; i < 2; ++i)
            #pragma unroll
            for (int j = 0; j < 4; ++j)
                acc[i][j] = __builtin_amdgcn_wmma_f32_16x16x32_f16(
                    false, a[i].v, false, bf[j].v, (short)0, acc[i][j],
                    false, false);
    }
    #pragma unroll
    for (int i = 0; i < 2; ++i) {
        #pragma unroll
        for (int r = 0; r < 8; ++r) {
            int m = tm * 32 + i * 16 + r + 8 * half;
            float bv = bias ? bias[m] : 0.0f;
            #pragma unroll
            for (int j = 0; j < 4; ++j)
                Cb[(long)m * Nn + tn * 64 + j * 16 + col] = acc[i][j][r] + bv;
        }
    }
}

// --------------------------- batchnorm stats -------------------------------
// One block per channel; biased stats over B*N samples. stats[c] = {mean, rstd}
__global__ __launch_bounds__(256) void bn_stats_kernel(
    const float* __restrict__ raw, float* __restrict__ stats, int C) {
    int c = blockIdx.x;
    float s = 0.f, q = 0.f;
    for (int idx = threadIdx.x; idx < BN_ * NP_; idx += 256) {
        int bb = idx / NP_, n = idx % NP_;
        float v = raw[((size_t)bb * C + c) * NP_ + n];
        s += v; q += v * v;
    }
    __shared__ float ss[256], sq[256];
    ss[threadIdx.x] = s; sq[threadIdx.x] = q;
    __syncthreads();
    for (int st = 128; st > 0; st >>= 1) {
        if (threadIdx.x < st) {
            ss[threadIdx.x] += ss[threadIdx.x + st];
            sq[threadIdx.x] += sq[threadIdx.x + st];
        }
        __syncthreads();
    }
    if (threadIdx.x == 0) {
        const float cnt = (float)(BN_ * NP_);
        float m = ss[0] / cnt;
        float var = sq[0] / cnt - m * m;
        stats[c * 2]     = m;
        stats[c * 2 + 1] = rsqrtf(var + 1e-5f);
    }
}

// normalize raw qkv, repack: q[b][h][n][dk], k[b][h][n][dk], vT[b][h][dv][n]
__global__ __launch_bounds__(256) void normalize_qkv_kernel(
    const float* __restrict__ raw, const float* __restrict__ stats,
    const float* __restrict__ gq, const float* __restrict__ bq,
    const float* __restrict__ gk, const float* __restrict__ bk,
    const float* __restrict__ gv, const float* __restrict__ bv,
    _Float16* __restrict__ qb, _Float16* __restrict__ kb,
    _Float16* __restrict__ vt) {
    size_t id = (size_t)blockIdx.x * blockDim.x + threadIdx.x;  // B*1024*N
    int n = id % NP_;
    int o = (id / NP_) % OQKV_;
    int b = id / ((size_t)NP_ * OQKV_);
    float m = stats[o * 2], rs = stats[o * 2 + 1];
    float v = (raw[id] - m) * rs;
    if (o < 256) {
        v = v * gq[o] + bq[o];
        int h = o >> 5, d = o & 31;
        qb[(((size_t)b * H_ + h) * NP_ + n) * DK_ + d] = (_Float16)v;
    } else if (o < 512) {
        int oo = o - 256;
        v = v * gk[oo] + bk[oo];
        int h = oo >> 5, d = oo & 31;
        kb[(((size_t)b * H_ + h) * NP_ + n) * DK_ + d] = (_Float16)v;
    } else {
        int oo = o - 512;
        v = v * gv[oo] + bv[oo];
        int h = oo >> 6, d = oo & 63;
        vt[(((size_t)b * H_ + h) * DV_ + d) * NP_ + n] = (_Float16)v;
    }
}

// --------------------------- fused flash attention -------------------------
// grid = B*H*8 blocks; 8 waves/block; each wave owns 16 query rows.
// S = q k^T * scale + bias ; online softmax ; acc += P v ; GELU ; gT f16 out.
__global__ __launch_bounds__(256) void attention_kernel(
    const _Float16* __restrict__ qb, const _Float16* __restrict__ kb,
    const _Float16* __restrict__ vt, const float* __restrict__ biasb,
    _Float16* __restrict__ gt) {
    const int lane = threadIdx.x & 31;
    const int half = lane >> 4, col = lane & 15;
    const int w = threadIdx.x >> 5;

    const int bh = blockIdx.x >> 3;          // b*H + h
    const int rblk = blockIdx.x & 7;
    const int b = bh / H_, h = bh % H_;
    const int i0 = rblk * 128 + w * 16;

    __shared__ _Float16 ptile[8][16][32];    // per-wave P tile (C->A restage)

    const _Float16* qbase = qb + ((size_t)bh * NP_) * DK_;
    const _Float16* kbase = kb + ((size_t)bh * NP_) * DK_;
    const _Float16* vbase = vt + ((size_t)bh * DV_) * NP_;
    const float*    bbase = biasb + (size_t)h * NP_ * NP_;

    // q A-fragment: rows i0..i0+15, K = 0..31 (whole head dim)
    FragU qa;
    {
        const _Float16* qr = qbase + (size_t)(i0 + col) * DK_;
        qa.h[0] = *(const v8h*)(qr + half * 16);
        qa.h[1] = *(const v8h*)(qr + half * 16 + 8);
    }

    v8f acc[4] = {};                // 16 x 64 output in f32
    float mrow[8], lrow[8];
    #pragma unroll
    for (int r = 0; r < 8; ++r) { mrow[r] = -1e30f; lrow[r] = 0.f; }

    const v8f zero8 = {};
    for (int jp = 0; jp < NP_; jp += 32) {
        // two k B-fragments (columns jp..jp+15, jp+16..jp+31)
        FragU kb0, kb1;
        {
            const _Float16* kr0 = kbase + (size_t)(jp + col) * DK_ + half * 16;
            kb0.h[0] = *(const v8h*)(kr0);
            kb0.h[1] = *(const v8h*)(kr0 + 8);
            const _Float16* kr1 = kbase + (size_t)(jp + 16 + col) * DK_ + half * 16;
            kb1.h[0] = *(const v8h*)(kr1);
            kb1.h[1] = *(const v8h*)(kr1 + 8);
        }
        v8f s0 = __builtin_amdgcn_wmma_f32_16x16x32_f16(
            false, qa.v, false, kb0.v, (short)0, zero8, false, false);
        v8f s1 = __builtin_amdgcn_wmma_f32_16x16x32_f16(
            false, qa.v, false, kb1.v, (short)0, zero8, false, false);

        float p0[8], p1[8];
        #pragma unroll
        for (int r = 0; r < 8; ++r) {
            int irow = i0 + r + 8 * half;
            const float* brw = bbase + (size_t)irow * NP_ + jp + col;
            float x0 = s0[r] * SCALE_ + brw[0];
            float x1 = s1[r] * SCALE_ + brw[16];
            float tmax = fmaxf(x0, x1);
            #pragma unroll
            for (int d = 1; d <= 8; d <<= 1)
                tmax = fmaxf(tmax, __shfl_xor(tmax, d, 32));
            float mnew = fmaxf(mrow[r], tmax);
            float corr = __expf(mrow[r] - mnew);
            float e0 = __expf(x0 - mnew);
            float e1 = __expf(x1 - mnew);
            float ps = e0 + e1;
            #pragma unroll
            for (int d = 1; d <= 8; d <<= 1)
                ps += __shfl_xor(ps, d, 32);
            lrow[r] = lrow[r] * corr + ps;
            mrow[r] = mnew;
            acc[0][r] *= corr; acc[1][r] *= corr;
            acc[2][r] *= corr; acc[3][r] *= corr;
            p0[r] = e0; p1[r] = e1;
        }

        // restage P (C layout) -> A layout via LDS
        #pragma unroll
        for (int r = 0; r < 8; ++r) {
            ptile[w][r + 8 * half][col]      = (_Float16)p0[r];
            ptile[w][r + 8 * half][16 + col] = (_Float16)p1[r];
        }
        __syncthreads();
        FragU pa;
        {
            const _Float16* pr = &ptile[w][col][0];
            pa.h[0] = *(const v8h*)(pr + half * 16);
            pa.h[1] = *(const v8h*)(pr + half * 16 + 8);
        }

        // acc[dt] += P (16x32) x v (32x16) for 4 dv tiles
        #pragma unroll
        for (int dt = 0; dt < 4; ++dt) {
            FragU vb;
            const _Float16* vr = vbase + (size_t)(dt * 16 + col) * NP_ + jp + half * 16;
            vb.h[0] = *(const v8h*)(vr);
            vb.h[1] = *(const v8h*)(vr + 8);
            acc[dt] = __builtin_amdgcn_wmma_f32_16x16x32_f16(
                false, pa.v, false, vb.v, (short)0, acc[dt], false, false);
        }
        __syncthreads();
    }

    // finalize softmax, GELU, store gT[b][n][h*64+d] f16
    #pragma unroll
    for (int dt = 0; dt < 4; ++dt) {
        #pragma unroll
        for (int r = 0; r < 8; ++r) {
            float o = acc[dt][r] / lrow[r];
            o = gelu_exact(o);
            int irow = i0 + r + 8 * half;
            int dcol = dt * 16 + col;
            gt[((size_t)b * NP_ + irow) * (H_ * DV_) + h * DV_ + dcol] = (_Float16)o;
        }
    }
}

// final BN on y[b][c][n] -> d_out
__global__ __launch_bounds__(256) void finalize_kernel(
    const float* __restrict__ y, const float* __restrict__ stats,
    const float* __restrict__ g, const float* __restrict__ be,
    float* __restrict__ out) {
    size_t id = (size_t)blockIdx.x * blockDim.x + threadIdx.x;  // B*256*N
    int c = (id / NP_) % DIM_;
    out[id] = (y[id] - stats[c * 2]) * stats[c * 2 + 1] * g[c] + be[c];
}

// ---------------------------------------------------------------------------

extern "C" void kernel_launch(void* const* d_in, const int* in_sizes, int n_in,
                              void* d_out, int out_size, void* d_ws, size_t ws_size,
                              hipStream_t stream) {
    (void)in_sizes; (void)n_in; (void)out_size; (void)ws_size;
    const float* x      = (const float*)d_in[0];
    const float* Wq     = (const float*)d_in[1];
    const float* gq     = (const float*)d_in[2];
    const float* bq     = (const float*)d_in[3];
    const float* Wk     = (const float*)d_in[4];
    const float* gk     = (const float*)d_in[5];
    const float* bk     = (const float*)d_in[6];
    const float* Wv     = (const float*)d_in[7];
    const float* gv     = (const float*)d_in[8];
    const float* bv     = (const float*)d_in[9];
    const float* Wo     = (const float*)d_in[10];
    const float* b_o    = (const float*)d_in[11];
    const float* go     = (const float*)d_in[12];
    const float* bo     = (const float*)d_in[13];
    const float* ptab   = (const float*)d_in[14];
    const int*   pidx   = (const int*)d_in[15];
    float* out = (float*)d_out;

    // workspace carve-up (256 B aligned)
    char* base = (char*)d_ws;
    size_t off = 0;
    auto carve = [&](size_t bytes) {
        void* p = base + off;
        off += (bytes + 255) & ~(size_t)255;
        return p;
    };
    _Float16* wqkv = (_Float16*)carve((size_t)OQKV_ * DIM_ * 2);           // 0.5 MB
    _Float16* wo   = (_Float16*)carve((size_t)DIM_ * H_ * DV_ * 2);        // 0.25 MB
    _Float16* xt   = (_Float16*)carve((size_t)BN_ * NP_ * DIM_ * 2);       // 16.8 MB
    float*    raw  = (float*)   carve((size_t)BN_ * OQKV_ * NP_ * 4);      // 134 MB
    float*    stq  = (float*)   carve((size_t)OQKV_ * 2 * 4);
    _Float16* qb   = (_Float16*)carve((size_t)BN_ * H_ * NP_ * DK_ * 2);   // 16.8 MB
    _Float16* kb   = (_Float16*)carve((size_t)BN_ * H_ * NP_ * DK_ * 2);   // 16.8 MB
    _Float16* vt   = (_Float16*)carve((size_t)BN_ * H_ * DV_ * NP_ * 2);   // 33.6 MB
    float*    bias = (float*)   carve((size_t)H_ * NP_ * NP_ * 4);         // 33.6 MB
    _Float16* gt   = (_Float16*)carve((size_t)BN_ * NP_ * H_ * DV_ * 2);   // 33.6 MB
    float*    yraw = (float*)   carve((size_t)BN_ * DIM_ * NP_ * 4);       // 33.6 MB
    float*    sto  = (float*)   carve((size_t)DIM_ * 2 * 4);

    // 1. weights -> f16
    convert_weights_kernel<<<1536, 256, 0, stream>>>(Wq, Wk, Wv, Wo, wqkv, wo);
    // 2. x -> xT f16
    transpose_x_kernel<<<32768, 256, 0, stream>>>(x, xt);
    // 3. QKV projection GEMM: [1024x256] x [256x1024] per batch
    //    macro-tiles: (1024/32)*(1024/64) = 512 per batch; 32*512/8 = 2048 blocks
    wmma_gemm_kernel<<<2048, 256, 0, stream>>>(
        wqkv, xt, raw, nullptr,
        OQKV_, NP_, DIM_, (long)NP_ * DIM_, (long)OQKV_ * NP_,
        (OQKV_ / 32) * (NP_ / 64));
    // 4. BN stats for 1024 channels
    bn_stats_kernel<<<OQKV_, 256, 0, stream>>>(raw, stq, OQKV_);
    // 5. normalize + repack q/k/vT f16
    normalize_qkv_kernel<<<131072, 256, 0, stream>>>(
        raw, stq, gq, bq, gk, bk, gv, bv, qb, kb, vt);
    // 6. relative position bias table [h][i][j]
    build_bias_kernel<<<32768, 256, 0, stream>>>(ptab, pidx, bias);
    // 7. fused flash attention + GELU -> gT f16
    attention_kernel<<<BN_ * H_ * 8, 256, 0, stream>>>(qb, kb, vt, bias, gt);
    // 8. output projection GEMM: [256x512] x [512x1024] per batch (+b_o)
    //    macro-tiles: (256/32)*(1024/64) = 128 per batch; 32*128/8 = 512 blocks
    wmma_gemm_kernel<<<512, 256, 0, stream>>>(
        wo, gt, yraw, b_o,
        DIM_, NP_, H_ * DV_, (long)NP_ * H_ * DV_, (long)DIM_ * NP_,
        (DIM_ / 32) * (NP_ / 64));
    // 9. BN stats for 256 output channels
    bn_stats_kernel<<<DIM_, 256, 0, stream>>>(yraw, sto, DIM_);
    // 10. final BN -> d_out
    finalize_kernel<<<32768, 256, 0, stream>>>(yraw, sto, go, bo, out);
}
